// LSTMPredictor_50483045597986
// MI455X (gfx1250) — compile-verified, matrix operand order checked by reading, not run on hardware
//
#include <hip/hip_runtime.h>

// ---------------------------------------------------------------------------
// 2-layer LSTM (B=1024, T=512, IN=19, H=128) + Linear(128->7), persistent
// kernel, bf16 WMMA (v_wmma_f32_16x16x32_bf16) with fp32 accumulation.
// Batch tiled 16 rows per workgroup (64 WGs); each WG sweeps all 512 steps.
// Cell states c1/c2 live in registers; layer-1 weights + W_lin are
// register-resident B-fragments; layer-2 weights staged in LDS (256 KB).
// Gate nonlinearities use native V_TANH_F32 when available.
// ---------------------------------------------------------------------------

typedef __attribute__((ext_vector_type(16))) __bf16 v16bf;
typedef __attribute__((ext_vector_type(8)))  float  v8f;

#define T_STEPS 512
#define IN_F    19
#define HID     128
#define OUT_F   7

__device__ __forceinline__ unsigned short f2bf(float f) {
  unsigned int u = __float_as_uint(f);
  u += 0x7FFFu + ((u >> 16) & 1u);   // round-to-nearest-even
  return (unsigned short)(u >> 16);
}

union FragCvt {
  uint4 q[2];
  v16bf b;
};

__device__ __forceinline__ v16bf ld_frag(const unsigned short* p) {
  FragCvt f;
  const uint4* q = (const uint4*)p;
  f.q[0] = q[0];
  f.q[1] = q[1];
  return f.b;
}

__device__ __forceinline__ v16bf zero_frag() {
  FragCvt f;
  f.q[0] = make_uint4(0u, 0u, 0u, 0u);
  f.q[1] = make_uint4(0u, 0u, 0u, 0u);
  return f.b;
}

__device__ __forceinline__ float fast_rcp(float x) {
#if __has_builtin(__builtin_amdgcn_rcpf)
  return __builtin_amdgcn_rcpf(x);
#else
  return 1.0f / x;
#endif
}

__device__ __forceinline__ float fast_tanh(float x) {
#if __has_builtin(__builtin_amdgcn_tanhf)
  return __builtin_amdgcn_tanhf(x);            // native V_TANH_F32
#elif __has_builtin(__builtin_amdgcn_tanh_f32)
  return __builtin_amdgcn_tanh_f32(x);
#else
  return 2.0f * fast_rcp(1.0f + __expf(-2.0f * x)) - 1.0f;  // NaN-safe
#endif
}

__device__ __forceinline__ float fast_sig(float x) {
#if __has_builtin(__builtin_amdgcn_tanhf) || __has_builtin(__builtin_amdgcn_tanh_f32)
  return fmaf(0.5f, fast_tanh(0.5f * x), 0.5f);
#else
  return fast_rcp(1.0f + __expf(-x));
#endif
}

__device__ __forceinline__ v8f wmma_bf16(v16bf a, v16bf b, v8f c) {
  return __builtin_amdgcn_wmma_f32_16x16x32_bf16(false, a, false, b, (short)0, c, false, false);
}

// Stage a 16x19 slab of x (padded to K=32 with zeros) into A-fragment layout.
__device__ __forceinline__ void stage_x(const float* __restrict__ x,
                                        unsigned short* xbuf,
                                        int b0, int t, int e0, int stride) {
  for (int e = e0; e < 16 * IN_F; e += stride) {
    int m = e / IN_F;
    int k = e - m * IN_F;
    float v = x[((size_t)(b0 + m) * T_STEPS + t) * IN_F + k];
    int ln = m + ((k & 8) ? 16 : 0);
    int h  = (k & 7) | ((k & 16) >> 1);
    xbuf[ln * 16 + h] = f2bf(v);
  }
}

// ---------------------------------------------------------------------------
// Prep kernel: fp32 weights -> bf16 WMMA B-fragments (ISA 16x16x32 lane
// layout) + combined biases.
//   frag ids: tile in [0,32), which in [0,13):
//     0 -> W_ih1 (K padded 19->32); 1..4 -> W_hh1; 5..8 -> W_ih2; 9..12 -> W_hh2
//   frag ids 416..419 -> W_lin (cols >= 7 zeroed)
// ---------------------------------------------------------------------------
__global__ void prep_fragments(const float* __restrict__ Wih1,
                               const float* __restrict__ Whh1,
                               const float* __restrict__ bih1,
                               const float* __restrict__ bhh1,
                               const float* __restrict__ Wih2,
                               const float* __restrict__ Whh2,
                               const float* __restrict__ bih2,
                               const float* __restrict__ bhh2,
                               const float* __restrict__ Wlin,
                               unsigned short* __restrict__ frag,
                               float* __restrict__ b1,
                               float* __restrict__ b2) {
  int blk = blockIdx.x;
  int t   = threadIdx.x;            // 0..511 = lane*16 + half
  if (blk < 416) {
    int tile  = blk / 13;
    int which = blk - tile * 13;
    int lane  = t >> 4;
    int h     = t & 15;
    int col   = tile * 16 + (lane & 15);        // gate output column (W row)
    int ko    = (h & 7) + ((h & 8) ? 16 : 0) + ((lane & 16) ? 8 : 0);
    float v;
    if (which == 0) {
      v = (ko < IN_F) ? Wih1[col * IN_F + ko] : 0.0f;
    } else if (which <= 4) {
      v = Whh1[col * HID + (which - 1) * 32 + ko];
    } else if (which <= 8) {
      v = Wih2[col * HID + (which - 5) * 32 + ko];
    } else {
      v = Whh2[col * HID + (which - 9) * 32 + ko];
    }
    frag[(size_t)blk * 512 + t] = f2bf(v);
  } else if (blk < 420) {
    int kb   = (blk - 416) * 32;
    int lane = t >> 4;
    int h    = t & 15;
    int c    = lane & 15;
    int ko   = (h & 7) + ((h & 8) ? 16 : 0) + ((lane & 16) ? 8 : 0);
    float v  = (c < OUT_F) ? Wlin[c * HID + kb + ko] : 0.0f;
    frag[(size_t)blk * 512 + t] = f2bf(v);
  } else if (blk == 420) {
    b1[t] = bih1[t] + bhh1[t];
  } else {
    b2[t] = bih2[t] + bhh2[t];
  }
}

// ---------------------------------------------------------------------------
// Persistent LSTM kernel: 64 blocks x 256 threads (8 wave32).
// Dynamic LDS layout (bytes):
//   [0, 256K)          layer-2 weight frags (32 KB per wave, private region)
//   [256K, +2K)        x fragment, double-buffered (2 x 1 KB)
//   [+2K, +6K)         h1 fragments (4 KB)
//   [+6K, +14K)        h2 fragments, double-buffered (2 x 4 KB)
// Total 276,480 B < 320 KB.
// ---------------------------------------------------------------------------
__global__ __launch_bounds__(256, 1) void lstm_persistent(
    const float* __restrict__ x,
    const unsigned short* __restrict__ frag,
    const float* __restrict__ b1v,
    const float* __restrict__ b2v,
    const float* __restrict__ blin,
    float* __restrict__ out) {
  extern __shared__ char smem[];
  unsigned short* w2buf = (unsigned short*)smem;
  unsigned short* xbuf  = (unsigned short*)(smem + 262144);            // 2 x 512 halves
  unsigned short* h1buf = (unsigned short*)(smem + 262144 + 2048);     // 2048 halves
  unsigned short* h2buf = (unsigned short*)(smem + 262144 + 2048 + 4096); // 2 x 2048

  const int tid  = threadIdx.x;
  const int lane = tid & 31;
  const int wave = tid >> 5;
  const int b0   = blockIdx.x * 16;
  const int col  = lane & 15;
  const int colInGate = 16 * wave + col;     // column owned within each gate

  // ---- layer-1 weight fragments + W_lin into registers ----
  v16bf wih1[4], whh1[4][4], wlin[4];
#pragma unroll
  for (int g = 0; g < 4; ++g) {
    const unsigned short* base = frag + (size_t)((8 * g + wave) * 13) * 512;
    wih1[g] = ld_frag(base + lane * 16);
#pragma unroll
    for (int k = 0; k < 4; ++k)
      whh1[g][k] = ld_frag(base + (1 + k) * 512 + lane * 16);
  }
#pragma unroll
  for (int k = 0; k < 4; ++k)
    wlin[k] = ld_frag(frag + (size_t)(416 + k) * 512 + lane * 16);

  // ---- layer-2 weight fragments into this wave's private LDS region ----
#pragma unroll 4
  for (int f = 0; f < 32; ++f) {
    int g = f >> 3, j = f & 7;
    const uint4* s = (const uint4*)(frag + (size_t)((8 * g + wave) * 13 + 5 + j) * 512 + lane * 16);
    uint4* d = (uint4*)(w2buf + wave * 16384 + f * 512 + lane * 16);
    d[0] = s[0];
    d[1] = s[1];
  }

  // ---- per-lane biases ----
  const float bi1 = b1v[colInGate],       bf1 = b1v[128 + colInGate];
  const float bg1 = b1v[256 + colInGate], bo1 = b1v[384 + colInGate];
  const float bi2 = b2v[colInGate],       bf2 = b2v[128 + colInGate];
  const float bg2 = b2v[256 + colInGate], bo2 = b2v[384 + colInGate];
  const float bl  = (col < OUT_F) ? blin[col] : 0.0f;

  // ---- zero state buffers (x pad + h2 initial state), stage x_0 ----
  for (int i = tid; i < 3584; i += 256)      // 14,336 bytes
    ((unsigned int*)(smem + 262144))[i] = 0u;
  __syncthreads();
  stage_x(x, xbuf, b0, 0, tid, 256);         // parity 0

  v8f c1 = (v8f)0.0f, c2 = (v8f)0.0f;
  v16bf a1[4];
#pragma unroll
  for (int k = 0; k < 4; ++k) a1[k] = zero_frag();
  __syncthreads();

  // A-fragment write position for the h value this lane produces
  const int kidx  = colInGate;
  const int fragk = kidx >> 5;
  const int ko    = kidx & 31;
  const int lnoff = (ko & 8) ? 16 : 0;
  const int hoff  = (ko & 7) | ((ko & 16) >> 1);
  const int rbase = (lane >> 4) * 8;         // accumulator row base

#pragma unroll 1
  for (int t = 0; t < T_STEPS; ++t) {
    unsigned short* xb_r = xbuf + (t & 1) * 512;
    unsigned short* xb_w = xbuf + ((t + 1) & 1) * 512;
    unsigned short* h2_r = h2buf + (t & 1) * 2048;        // h2_{t-1}
    unsigned short* h2_w = h2buf + ((t + 1) & 1) * 2048;  // h2_t

    // fragments for this step: x_t, h2_{t-1} (double-buffered, no barrier)
    v16bf xf = ld_frag(xb_r + lane * 16);
    v16bf a2[4];
#pragma unroll
    for (int k = 0; k < 4; ++k) a2[k] = ld_frag(h2_r + k * 512 + lane * 16);

    // ---- layer-1 gates: x_t@Wih1^T + h1_{t-1}@Whh1^T ----
    v8f acc[4];
#pragma unroll
    for (int g = 0; g < 4; ++g) {
      v8f a = (v8f)0.0f;
      a = wmma_bf16(xf, wih1[g], a);
#pragma unroll
      for (int k = 0; k < 4; ++k) a = wmma_bf16(a1[k], whh1[g][k], a);
      acc[g] = a;
    }

    // ---- layer-1 cell update (fp32), publish h1_t as bf16 A-frags ----
#pragma unroll
    for (int e = 0; e < 8; ++e) {
      float iv = fast_sig(acc[0][e] + bi1);
      float fv = fast_sig(acc[1][e] + bf1);
      float gv = fast_tanh(acc[2][e] + bg1);
      float ov = fast_sig(acc[3][e] + bo1);
      float cv = fv * c1[e] + iv * gv;
      c1[e] = cv;
      float hv = ov * fast_tanh(cv);
      int m = e + rbase;
      h1buf[fragk * 512 + (m + lnoff) * 16 + hoff] = f2bf(hv);
    }
    __syncthreads();   // B2: h1_t complete

#pragma unroll
    for (int k = 0; k < 4; ++k) a1[k] = ld_frag(h1buf + k * 512 + lane * 16);

    // ---- layer-2 gates: h1_t@Wih2^T + h2_{t-1}@Whh2^T (weights from LDS) ----
#pragma unroll
    for (int g = 0; g < 4; ++g) {
      v8f a = (v8f)0.0f;
      const unsigned short* wb = w2buf + wave * 16384 + g * 8 * 512;
#pragma unroll
      for (int k = 0; k < 4; ++k) a = wmma_bf16(a1[k], ld_frag(wb + k * 512 + lane * 16), a);
#pragma unroll
      for (int k = 0; k < 4; ++k) a = wmma_bf16(a2[k], ld_frag(wb + (4 + k) * 512 + lane * 16), a);
      acc[g] = a;
    }

    // ---- layer-2 cell update, publish h2_t ----
#pragma unroll
    for (int e = 0; e < 8; ++e) {
      float iv = fast_sig(acc[0][e] + bi2);
      float fv = fast_sig(acc[1][e] + bf2);
      float gv = fast_tanh(acc[2][e] + bg2);
      float ov = fast_sig(acc[3][e] + bo2);
      float cv = fv * c2[e] + iv * gv;
      c2[e] = cv;
      float hv = ov * fast_tanh(cv);
      int m = e + rbase;
      h2_w[fragk * 512 + (m + lnoff) * 16 + hoff] = f2bf(hv);
    }
    __syncthreads();   // B3: h2_t complete

    if (wave == 0) {
      // output Linear: y = h2_t @ Wlin^T + b (16-col tile, cols >= 7 padded)
      v16bf hh[4];
#pragma unroll
      for (int k = 0; k < 4; ++k) hh[k] = ld_frag(h2_w + k * 512 + lane * 16);
      v8f y = (v8f)0.0f;
#pragma unroll
      for (int k = 0; k < 4; ++k) y = wmma_bf16(hh[k], wlin[k], y);
      if (col < OUT_F) {
#pragma unroll
        for (int v = 0; v < 8; ++v)
          out[(size_t)(b0 + rbase + v) * (T_STEPS * OUT_F) + t * OUT_F + col] = y[v] + bl;
      }
    } else if (wave == 1 && t + 1 < T_STEPS) {
      stage_x(x, xb_w, b0, t + 1, lane, 32);   // prefetch next x fragment
    }
    __syncthreads();   // B4: xbuf/h2 ready for step t+1
  }
}

extern "C" void kernel_launch(void* const* d_in, const int* in_sizes, int n_in,
                              void* d_out, int out_size, void* d_ws, size_t ws_size,
                              hipStream_t stream) {
  (void)in_sizes; (void)n_in; (void)out_size; (void)ws_size;
  const float* x    = (const float*)d_in[0];
  const float* Wih1 = (const float*)d_in[1];
  const float* Whh1 = (const float*)d_in[2];
  const float* bih1 = (const float*)d_in[3];
  const float* bhh1 = (const float*)d_in[4];
  const float* Wih2 = (const float*)d_in[5];
  const float* Whh2 = (const float*)d_in[6];
  const float* bih2 = (const float*)d_in[7];
  const float* bhh2 = (const float*)d_in[8];
  const float* Wlin = (const float*)d_in[9];
  const float* blin = (const float*)d_in[10];

  // ws layout: 420 fragments x 1 KB, then b1[512], b2[512] (fp32)
  unsigned short* frag = (unsigned short*)d_ws;
  float* b1 = (float*)((char*)d_ws + 430080);
  float* b2 = (float*)((char*)d_ws + 432128);

  prep_fragments<<<422, 512, 0, stream>>>(Wih1, Whh1, bih1, bhh1,
                                          Wih2, Whh2, bih2, bhh2,
                                          Wlin, frag, b1, b2);

  const size_t smem = 262144 + 2048 + 4096 + 8192;   // 276,480 B < 320 KB
  lstm_persistent<<<64, 256, smem, stream>>>(x, frag, b1, b2, blin, (float*)d_out);
}
